// ViLSTM_75642964017440
// MI455X (gfx1250) — matrix operand order repaired and checked by reading, not compiled, not run on hardware
//
#include <hip/hip_runtime.h>
#include <hip/hip_bf16.h>

// ---------------------------------------------------------------------------
// ConvLSTM (ViLSTM) for MI455X / gfx1250.
//   T=20, B=8, Cin=256, hid=256, Cf=256, H=W=32.
//   Per step: gates = conv3x3([x_t;h], Wxh) + conv1x1(features,Wf) + b
//             then LSTM cell update with per-batch length mask.
// ~1.55 TFLOP of matmul -> bf16 WMMA (16x16x32, f32 accum) via implicit GEMM.
// Activations staged zero-padded (34x34) and channel-contiguous
// (Xt[b][pix][ci]) -> branch-free, fully vectorized staging for all 9 taps.
// Workgroup tile 128co x 128px; each wave32 computes 32x64 (8 WMMA/K-step,
// 1.5 ds_load_b128 per WMMA). bf16 weights (9.4MB) + acts (9.5MB) are
// L2-resident (192MB). Static ping-pong LDS, one barrier per K-step.
// ---------------------------------------------------------------------------

typedef __attribute__((ext_vector_type(16))) __bf16        v16bf;
typedef __attribute__((ext_vector_type(8)))  float          v8f;
typedef __attribute__((ext_vector_type(8)))  unsigned short u16x8;

#define T_STEPS 20
#define BATCH   8
#define CIN     256
#define HID     256
#define CF      256
#define CCAT    512            // Cin + hid
#define CO      1024           // 4*hid
#define HW      1024           // 32*32
#define WID     32
#define PW      34             // padded width (1px zero halo)
#define PHW     1156           // 34*34

// GEMM tiling
#define MT 128                 // out-channel tile
#define NT 128                 // pixel tile (4 rows of one image)
#define KC 32                  // ci chunk per WMMA K-step
#define BS_STRIDE 40           // Bs row stride in ushorts (80B, 16B-aligned)
#define NITER 144              // 9 taps * 16 ci-chunks (even -> clean 2x unroll)

__device__ __forceinline__ unsigned short f2bf(float x) {
    // round-to-nearest-even fp32 -> bf16
    unsigned u = __builtin_bit_cast(unsigned, x);
    u += 0x7FFFu + ((u >> 16) & 1u);
    return (unsigned short)(u >> 16);
}

// ---- one-time: repack Wxh fp32 [co][ci][3][3] -> bf16 [tap][co][ci] --------
__global__ void vilstm_repack(const float* __restrict__ Wxh,
                              unsigned short* __restrict__ Wp) {
    int tid = blockIdx.x * 256 + threadIdx.x;      // 9*1024*512 total
    int ci  = tid & 511;
    int co  = (tid >> 9) & 1023;
    int tap = tid >> 19;
    float v = Wxh[((size_t)co * CCAT + ci) * 9 + tap];
    Wp[tid] = f2bf(v);
}

// ---- one-time: feat_gates = conv1x1(features, Wf) + bias -------------------
__global__ void vilstm_fg(const float* __restrict__ feat,
                          const float* __restrict__ Wf,
                          const float* __restrict__ bias,
                          float* __restrict__ FG) {
    int tid = blockIdx.x * 256 + threadIdx.x;      // 8*1024*1024 total
    int p  = tid & (HW - 1);
    int co = (tid >> 10) & (CO - 1);
    int b  = tid >> 20;
    const float* f = feat + (size_t)b * CF * HW + p;
    const float* w = Wf + (size_t)co * CF;
    float s = bias[co];
#pragma unroll 4
    for (int cf = 0; cf < CF; ++cf) s += f[(size_t)cf * HW] * w[cf];
    FG[tid] = s;
}

// ---- per step: pack [x_t ; h_prev] fp32 -> bf16, padded + transposed -------
// Output layout: Xt[b][pp][ci], pp = padded pixel (34x34), ci contiguous.
__global__ void vilstm_pack(const float* __restrict__ x_t,
                            const float* __restrict__ h_prev,   // null at t=0
                            unsigned short* __restrict__ Xt) {
    int tid = blockIdx.x * 256 + threadIdx.x;      // 8*1156*512 total
    if (tid >= BATCH * PHW * CCAT) return;
    int ci = tid & 511;
    int bp = tid >> 9;                             // b*1156 + pp
    int pp = bp % PHW;
    int b  = bp / PHW;
    int y  = pp / PW - 1;
    int x  = pp % PW - 1;
    float v = 0.0f;
    if (y >= 0 && y < WID && x >= 0 && x < WID) {
        if (ci < CIN)    v = x_t[((size_t)b * CIN + ci) * HW + y * WID + x];
        else if (h_prev) v = h_prev[((size_t)b * HID + (ci - CIN)) * HW + y * WID + x];
    }
    Xt[tid] = f2bf(v);
}

// ---- per step: gates = conv3x3(concat, Wxh) + FG  (implicit GEMM, WMMA) ----
// Block 256 threads = 8 wave32. Tile M=128 co x N=128 px; wave tile 32x64.
// Static ping-pong LDS buffers, one barrier per K-step, branch-free staging.
__global__ __launch_bounds__(256)
void vilstm_gates_gemm(const unsigned short* __restrict__ Wp,   // [9][1024][512] bf16
                       const unsigned short* __restrict__ Xt,   // [8][1156][512] bf16
                       const float* __restrict__ FG,            // [8][1024][1024]
                       float* __restrict__ gates)               // [8][1024][1024]
{
    __shared__ unsigned short As0[MT * KC];          // 8KB each
    __shared__ unsigned short As1[MT * KC];
    __shared__ unsigned short Bs0[NT * BS_STRIDE];   // 10KB each
    __shared__ unsigned short Bs1[NT * BS_STRIDE];

    const int t   = threadIdx.x;
    const int co0 = blockIdx.x * MT;                 // 8 tiles over 1024 co
    const int pt  = blockIdx.y;                      // 64 pixel tiles
    const int bI  = pt >> 3;                         // image
    const int y0  = (pt & 7) << 2;                   // first of 4 output rows

    const int lane  = t & 31;
    const int w     = t >> 5;
    const int wm    = w & 3;                         // co sub-block of 32
    const int wn    = w >> 2;                        // pixel half of 64
    const int m     = lane & 15;
    const int khalf = lane >> 4;

    v8f acc[2][4] = {};

    // staging roles
    const int ar  = t >> 1;                          // A row 0..127
    const int ah  = (t & 1) << 4;                    // A ci sub-offset 0/16
    const int bq  = t & 127;                         // B pixel in tile
    const int bcg = (t >> 7) << 4;                   // B ci group 0/16
    const int by  = y0 + (bq >> 5);                  // output row of my pixel
    const int bx  = bq & 31;                         // output col of my pixel
    const int pixBase = by * PW + bx;                // padded pixel at tap(0,0)

    const unsigned short* wRow  = Wp + (size_t)(co0 + ar) * CCAT + ah;
    const unsigned short* xBase = Xt + (size_t)bI * PHW * CCAT + bcg;
    const int aOff = ar * KC + ah;
    const int bOff = bq * BS_STRIDE + bcg;

    // branch-free staging of pipeline stage `idx` into LDS buffers aD/bD
    auto stage = [&](int idx, unsigned short* aD, unsigned short* bD) {
        const int tap = idx >> 4;                    // 0..8
        const int kc  = (idx & 15) << 5;             // 0..480
        const int ty  = (tap * 11) >> 5;             // tap/3 for 0..8
        const int tx  = tap - ty * 3;
        // A: 16 contiguous bf16 per thread (2x global b128 -> 2x ds b128)
        const unsigned short* wsrc = wRow + (size_t)tap * (CO * CCAT) + kc;
        u16x8 a0 = *(const u16x8*)(wsrc);
        u16x8 a1 = *(const u16x8*)(wsrc + 8);
        __builtin_prefetch(wsrc + KC, 0, 1);
        *(u16x8*)(aD + aOff)     = a0;
        *(u16x8*)(aD + aOff + 8) = a1;
        // B: 16 contiguous bf16 per thread (2x global b128 -> 2x ds b128)
        const unsigned short* xsrc =
            xBase + (size_t)(pixBase + ty * PW + tx) * CCAT + kc;
        u16x8 b0 = *(const u16x8*)(xsrc);
        u16x8 b1 = *(const u16x8*)(xsrc + 8);
        *(u16x8*)(bD + bOff)     = b0;
        *(u16x8*)(bD + bOff + 8) = b1;
    };

    // fragment offsets (ushort units)
    const int arow0 = (wm * 32 + m) * KC + khalf * 16;       // A sub-tile 0
    const int arow1 = (wm * 32 + 16 + m) * KC + khalf * 16;  // A sub-tile 1
    int brow[4];
#pragma unroll
    for (int j = 0; j < 4; ++j)
        brow[j] = (wn * 64 + j * 16 + m) * BS_STRIDE + khalf * 16;

    auto compute = [&](const unsigned short* As_, const unsigned short* Bs_) {
        // A fragments (16x32 bf16): K = khalf*16 + e
        u16x8 a0lo = *(const u16x8*)(As_ + arow0);
        u16x8 a0hi = *(const u16x8*)(As_ + arow0 + 8);
        u16x8 a1lo = *(const u16x8*)(As_ + arow1);
        u16x8 a1hi = *(const u16x8*)(As_ + arow1 + 8);
        v16bf A0 = __builtin_bit_cast(v16bf, __builtin_shufflevector(
            a0lo, a0hi, 0,1,2,3,4,5,6,7,8,9,10,11,12,13,14,15));
        v16bf A1 = __builtin_bit_cast(v16bf, __builtin_shufflevector(
            a1lo, a1hi, 0,1,2,3,4,5,6,7,8,9,10,11,12,13,14,15));
        // B fragments (32x16 bf16): col = lane&15, K = khalf*16 + e
        v16bf Bf[4];
#pragma unroll
        for (int j = 0; j < 4; ++j) {
            u16x8 blo = *(const u16x8*)(Bs_ + brow[j]);
            u16x8 bhi = *(const u16x8*)(Bs_ + brow[j] + 8);
            Bf[j] = __builtin_bit_cast(v16bf, __builtin_shufflevector(
                blo, bhi, 0,1,2,3,4,5,6,7,8,9,10,11,12,13,14,15));
        }
#pragma unroll
        for (int j = 0; j < 4; ++j) {
            acc[0][j] = __builtin_amdgcn_wmma_f32_16x16x32_bf16(
                false, A0, false, Bf[j], (short)0, acc[0][j], false, false);
            acc[1][j] = __builtin_amdgcn_wmma_f32_16x16x32_bf16(
                false, A1, false, Bf[j], (short)0, acc[1][j], false, false);
        }
    };

    // software pipeline, 2x unrolled so all LDS addresses are static
    stage(0, As0, Bs0);
    for (int idx = 0; idx < NITER; idx += 2) {
        __syncthreads();                     // stage(idx) visible
        stage(idx + 1, As1, Bs1);            // idx+1 <= NITER-1 always (even NITER)
        compute(As0, Bs0);
        __syncthreads();                     // stage(idx+1) visible
        if (idx + 2 < NITER) stage(idx + 2, As0, Bs0);
        compute(As1, Bs1);
    }

    // epilogue: D layout -> lane col = lane&15, row = v + 8*khalf
#pragma unroll
    for (int i = 0; i < 2; ++i) {
#pragma unroll
        for (int j = 0; j < 4; ++j) {
            int q = wn * 64 + j * 16 + m;                    // pixel in tile
#pragma unroll
            for (int v = 0; v < 8; ++v) {
                int cog = co0 + wm * 32 + i * 16 + khalf * 8 + v;
                size_t idx = ((size_t)bI * CO + cog) * HW + (size_t)(y0 * WID + q);
                gates[idx] = acc[i][j][v] + FG[idx];
            }
        }
    }
}

// ---- per step: LSTM pointwise cell with length mask ------------------------
__global__ void vilstm_cell(const float* __restrict__ gates,
                            float* __restrict__ cbuf,
                            const float* __restrict__ h_prev,   // null at t=0
                            float* __restrict__ h_out,
                            const int* __restrict__ len, int t) {
    int tid = blockIdx.x * 256 + threadIdx.x;      // 8*256*1024 total
    int p = tid & 1023;
    int c = (tid >> 10) & 255;
    int b = tid >> 18;
    size_t gb = ((size_t)b * CO + c) * HW + p;
    float ig = gates[gb];
    float fg = gates[gb + (size_t)HID * HW];
    float gg = gates[gb + (size_t)2 * HID * HW];
    float og = gates[gb + (size_t)3 * HID * HW];
    float cp = h_prev ? cbuf[tid] : 0.0f;          // t=0: c0 = 0
    float hp = h_prev ? h_prev[tid] : 0.0f;
    float si = 1.0f / (1.0f + __expf(-ig));
    float sf = 1.0f / (1.0f + __expf(-fg));
    float so = 1.0f / (1.0f + __expf(-og));
    float tg = tanhf(gg);
    float cn = sf * cp + si * tg;
    float hn = so * tanhf(cn);
    bool act = t < len[b];
    h_out[tid] = act ? hn : hp;
    cbuf[tid]  = act ? cn : cp;
}

// ---------------------------------------------------------------------------
extern "C" void kernel_launch(void* const* d_in, const int* in_sizes, int n_in,
                              void* d_out, int out_size, void* d_ws, size_t ws_size,
                              hipStream_t stream) {
    const float* input_   = (const float*)d_in[0];   // (20,8,256,32,32)
    const float* features = (const float*)d_in[1];   // (8,256,32,32)
    const float* Wxh      = (const float*)d_in[2];   // (1024,512,3,3)
    const float* Wf       = (const float*)d_in[3];   // (1024,256,1,1)
    const float* bias     = (const float*)d_in[4];   // (1024,)
    const int*   length   = (const int*)d_in[5];     // (8,)

    // workspace layout (256B aligned)
    char* ws = (char*)d_ws;
    size_t off = 0;
    auto take = [&](size_t bytes) {
        void* p = ws + off;
        off += (bytes + 255) & ~(size_t)255;
        return p;
    };
    unsigned short* Wp    = (unsigned short*)take((size_t)9 * CO * CCAT * 2);    // 9.4 MB
    float*          FG    = (float*)take((size_t)BATCH * CO * HW * 4);           // 33.5 MB
    float*          gates = (float*)take((size_t)BATCH * CO * HW * 4);           // 33.5 MB
    unsigned short* Xt    = (unsigned short*)take((size_t)BATCH * PHW * CCAT*2); // 9.5 MB
    float*          cbuf  = (float*)take((size_t)BATCH * HID * HW * 4);          // 8.4 MB
    (void)ws_size; (void)in_sizes; (void)n_in; (void)out_size;

    float* hs = (float*)d_out;                     // (20,8,256,32,32) then h_n, c_n
    const size_t STEP = (size_t)BATCH * HID * HW;  // 2,097,152 (== x_t stride)

    vilstm_repack<<<(9 * CO * CCAT) / 256, 256, 0, stream>>>(Wxh, Wp);
    vilstm_fg<<<(BATCH * CO * HW) / 256, 256, 0, stream>>>(features, Wf, bias, FG);

    dim3 ggrid(CO / MT, (BATCH * HW) / NT);        // (8, 64)
    const int packN = (BATCH * PHW * CCAT + 255) / 256;
    for (int t = 0; t < T_STEPS; ++t) {
        const float* hprev = t ? hs + (size_t)(t - 1) * STEP : nullptr;
        vilstm_pack<<<packN, 256, 0, stream>>>(input_ + (size_t)t * STEP, hprev, Xt);
        vilstm_gates_gemm<<<ggrid, 256, 0, stream>>>(Wp, Xt, FG, gates);
        vilstm_cell<<<(int)(STEP / 256), 256, 0, stream>>>(
            gates, cbuf, hprev, hs + (size_t)t * STEP, length, t);
    }
    // h_n = hs[T-1] (mask already folded in), c_n = cbuf
    hipMemcpyAsync(hs + (size_t)T_STEPS * STEP, hs + (size_t)(T_STEPS - 1) * STEP,
                   STEP * sizeof(float), hipMemcpyDeviceToDevice, stream);
    hipMemcpyAsync(hs + (size_t)(T_STEPS + 1) * STEP, cbuf,
                   STEP * sizeof(float), hipMemcpyDeviceToDevice, stream);
}
